// BridgeShield_36747740184626
// MI455X (gfx1250) — matrix operand order repaired and checked by reading, not compiled, not run on hardware
//
#include <hip/hip_runtime.h>
#include <hip/hip_bf16.h>
#include <math.h>

// ---------------------------------------------------------------------------
// Types for CDNA5 WMMA / TDM builtins
// ---------------------------------------------------------------------------
typedef __bf16 v16bf __attribute__((ext_vector_type(16)));
typedef float  v8f   __attribute__((ext_vector_type(8)));
typedef unsigned int u32x4 __attribute__((ext_vector_type(4)));
typedef int          i32x8 __attribute__((ext_vector_type(8)));
typedef int          i32x4 __attribute__((ext_vector_type(4)));

#define HID 128
#define NHEAD 4
#define DHEAD 32
#define NEG_SLOPE 0.2f

__device__ __forceinline__ __bf16 f2bf(float f) { return (__bf16)f; }

__device__ __forceinline__ float lrelu(float x) {
  return x > 0.f ? x : NEG_SLOPE * x;
}

__device__ __forceinline__ void atomicMaxFloat(float* addr, float v) {
  if (v >= 0.f) atomicMax((int*)addr, __float_as_int(v));
  else          atomicMin((unsigned int*)addr, __float_as_uint(v));
}

// ---------------------------------------------------------------------------
// GEMM: C[M,128] = act(A[M,128] @ W[128,128] + bias)
// act: 0=none, 1=relu, 2=tanh, 3=tanh + per-column atomic sum into C[128]
//
// Block = 256 threads (8 waves) covering 128 rows. W is DMA'd into LDS by the
// Tensor Data Mover, then converted once to a transposed bf16 copy so each
// WMMA B fragment is one contiguous 32B LDS load. Each wave owns a 16-row
// stripe, loops over the 8 column tiles: 32 x v_wmma_f32_16x16x32_bf16.
// ---------------------------------------------------------------------------
__global__ __launch_bounds__(256)
void gemm128_bias_act(const float* __restrict__ A, const float* __restrict__ W,
                      const float* __restrict__ bias, float* __restrict__ C,
                      int M, int act) {
  __shared__ float  Wlds[HID * HID];   // 64 KB f32 staging (TDM target)
  __shared__ __bf16 Wt[HID * HID];     // 32 KB transposed bf16: Wt[n*128+k]

  // --- TDM: DMA the whole 128x128 fp32 weight tensor into LDS (wave 0) ---
  if (threadIdx.x < 32) {
    unsigned long long ga = (unsigned long long)(size_t)W;
    unsigned lds = (unsigned)(size_t)(void*)Wlds;  // low 32 bits = LDS offset
    u32x4 g0;
    g0[0] = 1u;                                    // count=1, user descriptor
    g0[1] = lds;                                   // lds_addr
    g0[2] = (unsigned)(ga & 0xFFFFFFFFu);          // global_addr[31:0]
    g0[3] = (unsigned)((ga >> 32) & 0x01FFFFFFu)   // global_addr[56:32]
          | (2u << 30);                            // type = 2 ("image")
    const unsigned dim0 = 128, dim1 = 128, tile0 = 128, tile1 = 128;
    const unsigned long long stride0 = 128;        // row stride (elements)
    i32x8 g1;
    g1[0] = (int)(2u << 16);                              // data_size = 4B
    g1[1] = (int)((dim0 & 0xFFFFu) << 16);                // tensor_dim0 lo16
    g1[2] = (int)(((dim0 >> 16) & 0xFFFFu) | ((dim1 & 0xFFFFu) << 16));
    g1[3] = (int)(((dim1 >> 16) & 0xFFFFu) | (tile0 << 16));  // tile_dim0
    g1[4] = (int)(tile1 & 0xFFFFu);                       // tile_dim1
    g1[5] = (int)(unsigned)(stride0 & 0xFFFFFFFFu);       // dim0_stride lo32
    g1[6] = (int)(unsigned)((stride0 >> 32) & 0xFFFFu);   // dim0_stride hi16
    g1[7] = 0;
    i32x4 g2 = {0, 0, 0, 0};
    i32x4 g3 = {0, 0, 0, 0};
    i32x8 g4 = {0, 0, 0, 0, 0, 0, 0, 0};
    __builtin_amdgcn_tensor_load_to_lds(g0, g1, g2, g3, g4, 0);
    __builtin_amdgcn_s_wait_tensorcnt(0);
  }
  __syncthreads();

  // --- one-time convert + transpose: f32 W[k][n] -> bf16 Wt[n][k] ---------
  {
    const int base = threadIdx.x * 64;
#pragma unroll 8
    for (int i = 0; i < 64; ++i) {
      const int e = base + i;
      const int k = e >> 7, n = e & 127;
      Wt[n * HID + k] = f2bf(Wlds[e]);
    }
  }
  __syncthreads();

  const int lane = threadIdx.x & 31;
  const int wave = threadIdx.x >> 5;
  const int lm   = lane & 15;
  const int lh   = lane >> 4;                    // lane half (0/1)
  const int row0 = blockIdx.x * 128 + wave * 16; // this wave's 16-row stripe

  int arow = row0 + lm;
  if (arow >= M) arow = M - 1;
  const float* __restrict__ Arow = A + (size_t)arow * HID;
  __builtin_prefetch(Arow, 0, 1);                // global_prefetch_b8

  // A fragments for all 4 K-steps (16-bit A 16x32 layout), converted once.
  v16bf afrag[4];
#pragma unroll
  for (int kk = 0; kk < 4; ++kk) {
    const int ka = kk * 32 + lh * 16;
#pragma unroll
    for (int j = 0; j < 8; ++j) {
      afrag[kk][j]     = f2bf(Arow[ka + j]);
      afrag[kk][j + 8] = f2bf(Arow[ka + 8 + j]);
    }
  }

  // Loop over the 8 column tiles; B fragment = one 32B contiguous LDS read.
#pragma unroll
  for (int nt = 0; nt < 8; ++nt) {
    const int nb = nt * 16 + lm;
    v8f acc = {};
#pragma unroll
    for (int kk = 0; kk < 4; ++kk) {
      const v16bf bv = *(const v16bf*)(Wt + nb * HID + kk * 32 + lh * 16);
      acc = __builtin_amdgcn_wmma_f32_16x16x32_bf16(
          false, afrag[kk], false, bv, (short)0, acc, false, false);
    }
    const int col = nt * 16 + lm;
    const float bval = bias[col];
    if (act == 3) {            // fused tanh + column-sum reduction
      float s = 0.f;
#pragma unroll
      for (int r = 0; r < 8; ++r) {
        const int m = row0 + r + 8 * lh;
        if (m < M) s += tanhf(acc[r] + bval);
      }
      atomicAdd(&C[col], s);
    } else {
#pragma unroll
      for (int r = 0; r < 8; ++r) {
        const int m = row0 + r + 8 * lh;
        if (m < M) {
          float v = acc[r] + bval;
          if (act == 1)      v = fmaxf(v, 0.f);
          else if (act == 2) v = tanhf(v);
          C[(size_t)m * HID + col] = v;
        }
      }
    }
  }
}

// ---------------------------------------------------------------------------
// Elementwise / attention helper kernels
// ---------------------------------------------------------------------------
__global__ void fill_kernel(float* __restrict__ p, float v, size_t n) {
  size_t i = (size_t)blockIdx.x * blockDim.x + threadIdx.x;
  if (i < n) p[i] = v;
}

__global__ void relu_kernel(float* __restrict__ p, size_t n) {
  size_t i = (size_t)blockIdx.x * blockDim.x + threadIdx.x;
  if (i < n) p[i] = fmaxf(p[i], 0.f);
}

// out[node*H+h] = dot(z[node, h*32 : h*32+32], att[h, :])
__global__ void node_att_dot(const float* __restrict__ z,
                             const float* __restrict__ att,
                             float* __restrict__ out, int n) {
  int i = blockIdx.x * blockDim.x + threadIdx.x;
  if (i >= n) return;
  int node = i >> 2, h = i & 3;
  const float* zp = z + (size_t)node * HID + h * DHEAD;
  const float* ap = att + h * DHEAD;
  float s = 0.f;
#pragma unroll
  for (int d = 0; d < DHEAD; ++d) s += zp[d] * ap[d];
  out[i] = s;
}

// pass 1: alpha -> ew[], segment max into m[] (dst-indexed)
__global__ void edge_max_kernel(const int* __restrict__ src, const int* __restrict__ dst,
                                const float* __restrict__ asrc, const float* __restrict__ adst,
                                float* __restrict__ m, float* __restrict__ ew, int n) {
  int i = blockIdx.x * blockDim.x + threadIdx.x;
  if (i >= n) return;
  int e = i >> 2, h = i & 3;
  int s = src[e], d = dst[e];
  float alpha = lrelu(asrc[s * NHEAD + h] + adst[d * NHEAD + h]);
  ew[i] = alpha;
  atomicMaxFloat(&m[d * NHEAD + h], alpha);
}

// pass 2: ew <- exp(alpha - m[dst]); segment sum into den[]
__global__ void edge_sum_kernel(const int* __restrict__ dst,
                                const float* __restrict__ m, float* __restrict__ den,
                                float* __restrict__ ew, int n) {
  int i = blockIdx.x * blockDim.x + threadIdx.x;
  if (i >= n) return;
  int e = i >> 2, h = i & 3;
  int d = dst[e];
  float ex = __expf(ew[i] - m[d * NHEAD + h]);
  ew[i] = ex;
  atomicAdd(&den[d * NHEAD + h], ex);
}

// pass 3: ew <- ex / den[dst]  (final softmax weight per edge,head)
__global__ void edge_w_kernel(const int* __restrict__ dst,
                              const float* __restrict__ den,
                              float* __restrict__ ew, int n) {
  int i = blockIdx.x * blockDim.x + threadIdx.x;
  if (i >= n) return;
  int e = i >> 2, h = i & 3;
  float dd = den[dst[e] * NHEAD + h];
  ew[i] = ew[i] / (dd > 0.f ? dd : 1.f);
}

// pass 4: agg[dst,f] += w[e,h(f)] * z[src,f]
__global__ void edge_agg_kernel(const int* __restrict__ src, const int* __restrict__ dst,
                                const float* __restrict__ ew,
                                const float* __restrict__ z, float* __restrict__ agg, int n) {
  int i = blockIdx.x * blockDim.x + threadIdx.x;
  if (i >= n) return;
  int e = i >> 7, f = i & 127, h = f >> 5;
  int s = src[e], d = dst[e];
  float w = ew[e * NHEAD + h];
  atomicAdd(&agg[(size_t)d * HID + f], z[(size_t)s * HID + f] * w);
}

// scores r = dot(q, colsum[r]) / M ; softmax over R -> attnw
__global__ void semantic_softmax_kernel(const float* __restrict__ q,
                                        const float* __restrict__ colsum,
                                        float* __restrict__ attnw, int R, int M) {
  __shared__ float sh[128];
  __shared__ float sc[4];
  for (int r = 0; r < R; ++r) {
    sh[threadIdx.x] = q[threadIdx.x] * colsum[r * HID + threadIdx.x];
    __syncthreads();
    for (int st = 64; st > 0; st >>= 1) {
      if ((int)threadIdx.x < st) sh[threadIdx.x] += sh[threadIdx.x + st];
      __syncthreads();
    }
    if (threadIdx.x == 0) sc[r] = sh[0] / (float)M;
    __syncthreads();
  }
  if (threadIdx.x == 0) {
    float mx = -1e30f;
    for (int r = 0; r < R; ++r) mx = fmaxf(mx, sc[r]);
    float s = 0.f;
    for (int r = 0; r < R; ++r) { sc[r] = __expf(sc[r] - mx); s += sc[r]; }
    for (int r = 0; r < R; ++r) attnw[r] = sc[r] / s;
  }
}

__global__ void combine_kernel(const float* __restrict__ a0, const float* __restrict__ a1,
                               const float* __restrict__ attnw, float* __restrict__ out,
                               size_t n, int R) {
  size_t i = (size_t)blockIdx.x * blockDim.x + threadIdx.x;
  if (i >= n) return;
  float v = attnw[0] * a0[i];
  if (R > 1) v += attnw[1] * a1[i];
  out[i] = v;
}

__global__ void pool_max_kernel(const float* __restrict__ x, const int* __restrict__ batch,
                                float* __restrict__ gm, size_t n) {
  size_t i = (size_t)blockIdx.x * blockDim.x + threadIdx.x;
  if (i >= n) return;
  int node = (int)(i >> 7), f = (int)(i & 127);
  atomicMaxFloat(&gm[(size_t)batch[node] * HID + f], x[i]);
}

__global__ void head_out_kernel(const float* __restrict__ gm, const float* __restrict__ Wo,
                                const float* __restrict__ bo, float* __restrict__ out,
                                int G, int OUTC) {
  int i = blockIdx.x * blockDim.x + threadIdx.x;
  if (i >= G * OUTC) return;
  int g = i / OUTC, o = i % OUTC;
  float s = bo[o];
  for (int f = 0; f < HID; ++f) {
    float v = gm[(size_t)g * HID + f];
    if (!(v > -1e37f && v < 1e37f)) v = 0.f;   // guard empty graphs (-inf)
    s += v * Wo[f * OUTC + o];
  }
  out[i] = s;
}

// ---------------------------------------------------------------------------
// Host orchestration
// ---------------------------------------------------------------------------
static inline unsigned blocks_for(size_t n, int t) { return (unsigned)((n + t - 1) / t); }

extern "C" void kernel_launch(void* const* d_in, const int* in_sizes, int n_in,
                              void* d_out, int out_size, void* d_ws, size_t ws_size,
                              hipStream_t stream) {
  const int Nn = in_sizes[0] / HID;      // 100000
  const int Ee = in_sizes[3] / 2;        // 500000
  const int Gg = 64, OUTC = 8;

  // ---- input index map ----------------------------------------------------
  // 0..2: x_a,x_b,x_c   3..6: edge_ab,ba,bc,cb   7..9: batch_a,b,c
  // params flattened in jax sorted-key pytree order:
  //  layers[l] (base LB): att_dst{ab,ba,bc,cb}=+0..3, att_src=+4..7,
  //    k_lin.W=+8, k_lin.b=+9, proj {a,b,c}.{W,b}=+10..15, q=+16
  //  lin_out.W=44, lin_out.b=45
  //  mlp_t (t=0..2): base 46+4t: lin1.W,lin1.b,lin2.W,lin2.b
  auto P = [&](int i) { return (const float*)d_in[i]; };
  const int LB[2] = {10, 27};

  const int relSrc[4] = {0, 1, 1, 2};
  const int relDst[4] = {1, 0, 2, 1};
  const int* relEdge[4] = {(const int*)d_in[3], (const int*)d_in[4],
                           (const int*)d_in[5], (const int*)d_in[6]};
  const int typeRel[3][2] = {{1, -1}, {0, 3}, {2, -1}};  // relations into each dst type
  const int typeR[3] = {1, 2, 1};

  // ---- workspace layout (floats) -----------------------------------------
  float* ws = (float*)d_ws;
  const size_t NH = (size_t)Nn * HID;
  const size_t H4 = (size_t)Nn * NHEAD;
  const size_t EH = (size_t)Ee * NHEAD;
  float* X[3]   = {ws, ws + NH, ws + 2 * NH};
  float* Z[3]   = {ws + 3 * NH, ws + 4 * NH, ws + 5 * NH};
  float* TMP    = ws + 6 * NH;
  float* AGG[4] = {ws + 7 * NH, ws + 8 * NH, ws + 9 * NH, ws + 10 * NH};
  float* s0 = ws + 11 * NH;
  float* ASRC  = s0;           s0 += H4;
  float* ADST  = s0;           s0 += H4;
  float* MBUF  = s0;           s0 += H4;
  float* DEN   = s0;           s0 += H4;
  float* EW    = s0;           s0 += EH;           // per-edge alpha/exp/weight
  float* MEANS = s0;           s0 += 2 * HID;
  float* ATTW  = s0;           s0 += 4;
  float* GM    = s0;           s0 += (size_t)Gg * HID;

  const dim3 gblk((unsigned)((Nn + 127) / 128));
  auto gemm = [&](const float* A, const float* W, const float* b, float* C, int act) {
    gemm128_bias_act<<<gblk, 256, 0, stream>>>(A, W, b, C, Nn, act);
  };

  // ---- 1. per-type MLP + relu --------------------------------------------
  for (int t = 0; t < 3; ++t) {
    const int mb = 46 + 4 * t;
    gemm((const float*)d_in[t], P(mb + 0), P(mb + 1), TMP, 1);
    gemm(TMP, P(mb + 2), P(mb + 3), X[t], 1);
  }

  // ---- 2. HAN layers ------------------------------------------------------
  for (int l = 0; l < 2; ++l) {
    const int B = LB[l];
    // per-type projection to Z
    for (int t = 0; t < 3; ++t)
      gemm(X[t], P(B + 10 + 2 * t), P(B + 11 + 2 * t), Z[t], 0);

    // per-relation edge attention + aggregation
    for (int r = 0; r < 4; ++r) {
      const int s = relSrc[r], d = relDst[r];
      const int* srcp = relEdge[r];
      const int* dstp = relEdge[r] + Ee;
      const int nNH = Nn * NHEAD, nEH = Ee * NHEAD, nEF = Ee * HID;

      node_att_dot<<<blocks_for(nNH, 256), 256, 0, stream>>>(Z[s], P(B + 4 + r), ASRC, nNH);
      node_att_dot<<<blocks_for(nNH, 256), 256, 0, stream>>>(Z[d], P(B + 0 + r), ADST, nNH);
      fill_kernel<<<blocks_for(H4, 256), 256, 0, stream>>>(MBUF, -INFINITY, H4);
      fill_kernel<<<blocks_for(H4, 256), 256, 0, stream>>>(DEN, 0.f, H4);
      fill_kernel<<<blocks_for(NH, 256), 256, 0, stream>>>(AGG[r], 0.f, NH);
      edge_max_kernel<<<blocks_for(nEH, 256), 256, 0, stream>>>(srcp, dstp, ASRC, ADST,
                                                                MBUF, EW, nEH);
      edge_sum_kernel<<<blocks_for(nEH, 256), 256, 0, stream>>>(dstp, MBUF, DEN, EW, nEH);
      edge_w_kernel<<<blocks_for(nEH, 256), 256, 0, stream>>>(dstp, DEN, EW, nEH);
      edge_agg_kernel<<<blocks_for(nEF, 256), 256, 0, stream>>>(srcp, dstp, EW,
                                                                Z[s], AGG[r], nEF);
      relu_kernel<<<blocks_for(NH, 256), 256, 0, stream>>>(AGG[r], NH);
    }

    // semantic (relation-level) attention per dst type
    for (int t = 0; t < 3; ++t) {
      const int R = typeR[t];
      for (int i = 0; i < R; ++i) {
        const int r = typeRel[t][i];
        fill_kernel<<<1, 128, 0, stream>>>(MEANS + i * HID, 0.f, HID);
        gemm(AGG[r], P(B + 8), P(B + 9), MEANS + i * HID, 3);  // fused tanh+colsum
      }
      semantic_softmax_kernel<<<1, 128, 0, stream>>>(P(B + 16), MEANS, ATTW, R, Nn);
      const float* a0 = AGG[typeRel[t][0]];
      const float* a1 = (R > 1) ? AGG[typeRel[t][1]] : a0;
      combine_kernel<<<blocks_for(NH, 256), 256, 0, stream>>>(a0, a1, ATTW, X[t], NH, R);
    }
  }

  // ---- 3. global max pool + classifier -----------------------------------
  fill_kernel<<<blocks_for((size_t)Gg * HID, 256), 256, 0, stream>>>(GM, -INFINITY, (size_t)Gg * HID);
  for (int t = 0; t < 3; ++t)
    pool_max_kernel<<<blocks_for(NH, 256), 256, 0, stream>>>(X[t], (const int*)d_in[7 + t], GM, NH);
  head_out_kernel<<<blocks_for(Gg * OUTC, 256), 256, 0, stream>>>(GM, P(44), P(45),
                                                                 (float*)d_out, Gg, OUTC);
}
